// PromptGeneratorModel_29695403884668
// MI455X (gfx1250) — matrix operand order repaired and checked by reading, not compiled, not run on hardware
//
#include <hip/hip_runtime.h>

// ---------------- problem constants (from reference) ----------------
#define NUM_BEAMS 4
#define MAX_LEN   32
#define EOS       2
#define REP_PEN   1.2f
#define VOCAB     50257
#define D_MODEL   512
#define BATCH     32
#define WTILES    4                     // N tiles per wave
#define BLK_NT    (8 * WTILES)          // 32 N tiles per 8-wave block
#define VPAD      50688                 // 512 * 99 (vocab padded to block multiple)
#define NTILES    3168                  // VPAD / 16
#define KTILES    16                    // D_MODEL / 32
#define ROWS_MAX  (BATCH * NUM_BEAMS)   // 128

typedef __bf16 bf16;
typedef bf16  bf16x8  __attribute__((ext_vector_type(8)));
typedef bf16  bf16x16 __attribute__((ext_vector_type(16)));
typedef float f32x8   __attribute__((ext_vector_type(8)));
typedef unsigned int u32x4 __attribute__((ext_vector_type(4)));
typedef int   i32x4  __attribute__((ext_vector_type(4)));
typedef int   i32x8  __attribute__((ext_vector_type(8)));

// ---------------------------------------------------------------
// Pack proj [512][50257] f32 -> bf16 tiles laid out exactly as the
// WMMA B-fragment wants them:
//   tile (kk, nt): 32 lanes (N = lane & 15, K half = lane >> 4) x 16 K values,
//   1024 contiguous bytes per tile -> fully coalesced wave loads.
// ---------------------------------------------------------------
__global__ void pack_proj_kernel(const float* __restrict__ proj,
                                 bf16* __restrict__ projp) {
    long i = (long)blockIdx.x * blockDim.x + threadIdx.x;
    const long total = (long)KTILES * NTILES * 32 * 16;
    if (i >= total) return;
    int  e    = (int)(i & 15);         // fragment element (K, table order)
    int  lane = (int)((i >> 4) & 31);  // N = lane & 15, K half = lane >> 4
    long tile = i >> 9;                // kk * NTILES + nt
    int  nt   = (int)(tile % NTILES);
    int  kk   = (int)(tile / NTILES);
    int  k    = kk * 32 + ((lane >> 4) << 3) + (e < 8 ? e : 16 + (e - 8));
    int  n    = nt * 16 + (lane & 15);
    float v = (n < VOCAB) ? proj[(long)k * VOCAB + n] : 0.0f;
    projp[i] = (bf16)v;
}

// ---------------------------------------------------------------
// Mean-pool prefix embeddings -> h[row][512] in bf16.
// ---------------------------------------------------------------
__global__ void pool_embed_kernel(const float* __restrict__ emb,
                                  const int* __restrict__ toks,
                                  int tokStride, int curLen,
                                  bf16* __restrict__ hbf) {
    int row = blockIdx.x;
    int tid = threadIdx.x;             // 128 threads, 4 dims each (coalesced)
    float acc0 = 0.f, acc1 = 0.f, acc2 = 0.f, acc3 = 0.f;
    for (int l = 0; l < curLen; ++l) {
        int tok = toks[row * tokStride + l];
        const float* er = emb + (long)tok * D_MODEL;
        acc0 += er[tid];
        acc1 += er[tid + 128];
        acc2 += er[tid + 256];
        acc3 += er[tid + 384];
    }
    float inv = 1.0f / (float)curLen;
    bf16* hr = hbf + (long)row * D_MODEL;
    hr[tid]       = (bf16)(acc0 * inv);
    hr[tid + 128] = (bf16)(acc1 * inv);
    hr[tid + 256] = (bf16)(acc2 * inv);
    hr[tid + 384] = (bf16)(acc3 * inv);
}

// ---------------------------------------------------------------
// GEMM: logits[M, VPAD] = h[M,512](bf16) x proj(bf16 tiles), f32 acc.
// Block = 8 waves; each wave owns FOUR 16x16 tiles (4 independent
// accumulators, 4x A-fragment reuse). A tile (16x512 bf16 = 16KB)
// staged into LDS by the Tensor Data Mover (one 2KB slice per wave).
// K-loop is software-pipelined: next iteration's A (LDS) and B
// (global) fragments are issued before the current 4 WMMAs, so the
// loads overlap matrix math instead of draining loadcnt to zero.
// ---------------------------------------------------------------
__global__ __launch_bounds__(256)
void gemm_wmma_kernel(const bf16* __restrict__ hbf,
                      const bf16* __restrict__ projp,
                      float* __restrict__ logits) {
    __shared__ __align__(16) bf16 As[16 * D_MODEL];   // 16KB

    const int tid   = threadIdx.x;
    const int mTile = blockIdx.y;

    // ---- TDM: async-copy this block's A tile (8 waves x 2KB slices) ----
    {
        // wave index as an SGPR (descriptor operands must be scalar)
        int wv = __builtin_amdgcn_readfirstlane(tid >> 5);
        unsigned long long ga =
            (unsigned long long)(const void*)(hbf + (long)mTile * 16 * D_MODEL)
            + (unsigned long long)wv * 2048ull;
        // generic LDS pointer: low 32 bits are the wave-relative LDS offset
        unsigned ldsA = (unsigned)(unsigned long long)(void*)As + (unsigned)wv * 2048u;

        // D# group 0: count=1 | lds_addr | global_addr[56:0] | type=2
        u32x4 g0;
        g0[0] = 1u;
        g0[1] = ldsA;
        g0[2] = (unsigned)(ga & 0xffffffffull);
        g0[3] = (unsigned)((ga >> 32) & 0x01ffffffull) | (2u << 30);
        // D# group 1: 1-D tile of 1024 2-byte elements
        i32x8 g1;
        g1[0] = 0x00010000;        // data_size = 1 (2 bytes); no mask/pad/barrier
        g1[1] = (int)(1024u << 16);// tensor_dim0[15:0]=1024 (upper half)
        g1[2] = 0x00010000;        // tensor_dim0[31:16]=0 | tensor_dim1[15:0]=1
        g1[3] = (int)(1024u << 16);// tensor_dim1[31:16]=0 | tile_dim0=1024
        g1[4] = 0;                 // tile_dim1=0, tile_dim2=0 (1-D)
        g1[5] = 1024;              // tensor_dim0_stride[31:0]
        g1[6] = 0;
        g1[7] = 0;
        i32x4 z4 = {0, 0, 0, 0};
#if defined(__clang_major__) && (__clang_major__ >= 23)
        i32x8 z8 = {0, 0, 0, 0, 0, 0, 0, 0};
        __builtin_amdgcn_tensor_load_to_lds(g0, g1, z4, z4, z8, 0);
#else
        __builtin_amdgcn_tensor_load_to_lds(g0, g1, z4, z4, 0);
#endif
        __builtin_amdgcn_s_wait_tensorcnt(0);
    }
    __syncthreads();

    const int lane   = tid & 31;
    const int wave   = tid >> 5;
    const int nTile0 = (blockIdx.x * 8 + wave) * WTILES;  // 4 consecutive tiles
    const int m      = lane & 15;                // A: lane = M
    const int kb     = (lane >> 4) * 8;          // A: K sub-offset per lane half

    const bf16* bbase = projp + (long)nTile0 * 512 + (long)lane * 16;
    const long  bstep = (long)NTILES * 512;      // advance one K-chunk

    // A fragment per ISA 16-bit A 16x32 layout:
    // elems 0..7 = K[k0+kb .. +8), elems 8..15 = K[k0+kb+16 .. +24)
    auto loadA = [&](int kk) -> bf16x16 {
        const int k0 = kk * 32;
        bf16x8 lo = *(const bf16x8*)&As[m * D_MODEL + k0 + kb];
        bf16x8 hi = *(const bf16x8*)&As[m * D_MODEL + k0 + kb + 16];
        bf16x16 a;
        #pragma unroll
        for (int j = 0; j < 8; ++j) { a[j] = lo[j]; a[j + 8] = hi[j]; }
        return a;
    };

    f32x8 acc0 = {}, acc1 = {}, acc2 = {}, acc3 = {};

    // ---- prologue: fragments for kk = 0 ----
    bf16x16 aC = loadA(0);
    bf16x16 b0 = *(const bf16x16*)(bbase);
    bf16x16 b1 = *(const bf16x16*)(bbase + 512);
    bf16x16 b2 = *(const bf16x16*)(bbase + 1024);
    bf16x16 b3 = *(const bf16x16*)(bbase + 1536);

    // ---- pipelined main loop ----
    for (int kk = 0; kk < KTILES - 1; ++kk) {
        const bf16* np = bbase + (long)(kk + 1) * bstep;
        bf16x16 n0 = *(const bf16x16*)(np);
        bf16x16 n1 = *(const bf16x16*)(np + 512);
        bf16x16 n2 = *(const bf16x16*)(np + 1024);
        bf16x16 n3 = *(const bf16x16*)(np + 1536);
        bf16x16 aN = loadA(kk + 1);
        if (kk + 2 < KTILES)
            __builtin_prefetch(bbase + (long)(kk + 2) * bstep, 0, 0);

        acc0 = __builtin_amdgcn_wmma_f32_16x16x32_bf16(false, aC, false, b0, (short)0, acc0, false, false);
        acc1 = __builtin_amdgcn_wmma_f32_16x16x32_bf16(false, aC, false, b1, (short)0, acc1, false, false);
        acc2 = __builtin_amdgcn_wmma_f32_16x16x32_bf16(false, aC, false, b2, (short)0, acc2, false, false);
        acc3 = __builtin_amdgcn_wmma_f32_16x16x32_bf16(false, aC, false, b3, (short)0, acc3, false, false);

        aC = aN; b0 = n0; b1 = n1; b2 = n2; b3 = n3;
    }
    // ---- epilogue: last K-chunk ----
    acc0 = __builtin_amdgcn_wmma_f32_16x16x32_bf16(false, aC, false, b0, (short)0, acc0, false, false);
    acc1 = __builtin_amdgcn_wmma_f32_16x16x32_bf16(false, aC, false, b1, (short)0, acc1, false, false);
    acc2 = __builtin_amdgcn_wmma_f32_16x16x32_bf16(false, aC, false, b2, (short)0, acc2, false, false);
    acc3 = __builtin_amdgcn_wmma_f32_16x16x32_bf16(false, aC, false, b3, (short)0, acc3, false, false);

    // C/D layout: VGPR r -> M = r (lanes 0-15) / 8+r (lanes 16-31); N = lane&15
    const int rowBase = mTile * 16 + ((lane >> 4) << 3);
    const int colBase = nTile0 * 16 + (lane & 15);
    #pragma unroll
    for (int r = 0; r < 8; ++r) {
        float* orow = logits + (long)(rowBase + r) * VPAD + colBase;
        orow[0]  = acc0[r];
        orow[16] = acc1[r];
        orow[32] = acc2[r];
        orow[48] = acc3[r];
    }
}

// ---------------------------------------------------------------
// Repetition penalty (gather originals, then scatter) + forced EOS.
// ---------------------------------------------------------------
__global__ void rep_penalty_kernel(float* __restrict__ logits,
                                   const int* __restrict__ tokbuf,
                                   int curLen, int nRows, int forceEos) {
    int row = blockIdx.x * blockDim.x + threadIdx.x;
    if (row >= nRows) return;
    float g[MAX_LEN];
    int   tk[MAX_LEN];
    float* lr = logits + (long)row * VPAD;
    for (int l = 0; l < curLen; ++l) {
        tk[l] = tokbuf[row * MAX_LEN + l];
        g[l]  = lr[tk[l]];
    }
    for (int l = 0; l < curLen; ++l) {
        float v = g[l];
        lr[tk[l]] = (v < 0.f) ? v * REP_PEN : v / REP_PEN;
    }
    if (forceEos) lr[EOS] += 1e32f;
}

// ---------------------------------------------------------------
// Fused softmax-normalizer + top-8 in a single pass over the row:
// top-k is invariant under adding a per-row constant, so select on
// raw logits while accumulating online (max, sumexp); adjust the 8
// winners by (beam_score - lse) at the very end.
// ---------------------------------------------------------------
__global__ __launch_bounds__(256)
void row_softtop_kernel(const float* __restrict__ logits,
                        const float* __restrict__ bscore,   // null on step 1
                        float* __restrict__ topv, int* __restrict__ topi) {
    __shared__ float sv[256 * 8];
    __shared__ int   si[256 * 8];
    __shared__ float sm[256], ss[256];
    const int row = blockIdx.x;
    const float* lr = logits + (long)row * VPAD;

    float bv[8]; int bi[8];
    #pragma unroll
    for (int j = 0; j < 8; ++j) { bv[j] = -3.4e38f; bi[j] = 0; }
    float m = -3.4e38f, s = 0.f;
    for (int v = threadIdx.x; v < VOCAB; v += 256) {
        float x = lr[v];
        // online logsumexp
        if (x > m) { s = s * __expf(m - x) + 1.f; m = x; }
        else       { s += __expf(x - m); }
        // per-thread sorted top-8 (ties keep the earlier/lower index)
        if (x > bv[7]) {
            int j = 7;
            while (j > 0 && bv[j - 1] < x) { bv[j] = bv[j - 1]; bi[j] = bi[j - 1]; --j; }
            bv[j] = x; bi[j] = v;
        }
    }
    #pragma unroll
    for (int j = 0; j < 8; ++j) { sv[threadIdx.x * 8 + j] = bv[j]; si[threadIdx.x * 8 + j] = bi[j]; }
    sm[threadIdx.x] = m; ss[threadIdx.x] = s;
    __syncthreads();
    for (int off = 128; off > 0; off >>= 1) {
        if ((int)threadIdx.x < off) {
            // merge the two sorted 8-lists
            float* av  = &sv[threadIdx.x * 8];          int* ai  = &si[threadIdx.x * 8];
            float* bvp = &sv[(threadIdx.x + off) * 8];  int* bip = &si[(threadIdx.x + off) * 8];
            float mv[8]; int mi[8];
            int p = 0, q = 0;
            #pragma unroll
            for (int j = 0; j < 8; ++j) {
                if (av[p] >= bvp[q]) { mv[j] = av[p]; mi[j] = ai[p]; ++p; }
                else                 { mv[j] = bvp[q]; mi[j] = bip[q]; ++q; }
            }
            #pragma unroll
            for (int j = 0; j < 8; ++j) { av[j] = mv[j]; ai[j] = mi[j]; }
            // combine (max, sumexp) pairs
            float m1 = sm[threadIdx.x], s1 = ss[threadIdx.x];
            float m2 = sm[threadIdx.x + off], s2 = ss[threadIdx.x + off];
            float M = fmaxf(m1, m2);
            sm[threadIdx.x] = M;
            ss[threadIdx.x] = s1 * __expf(m1 - M) + s2 * __expf(m2 - M);
        }
        __syncthreads();
    }
    if (threadIdx.x < 8) {
        float lse  = sm[0] + __logf(ss[0]);
        float base = (bscore ? bscore[row] : 0.f) - lse;
        topv[row * 8 + threadIdx.x] = sv[threadIdx.x] + base;
        topi[row * 8 + threadIdx.x] = si[threadIdx.x];
    }
}

// ---------------------------------------------------------------
// First decode step: from per-batch top-8, emulate topk(K+1)=5 then
// keep first K non-EOS; init token buffer + beam scores.
// ---------------------------------------------------------------
__global__ void beam_init_kernel(const float* __restrict__ topv,
                                 const int* __restrict__ topi,
                                 const int* __restrict__ tokens0,
                                 int* __restrict__ tokOut,
                                 float* __restrict__ beamScores) {
    __shared__ int   selTok[NUM_BEAMS];
    __shared__ float selSc[NUM_BEAMS];
    const int b = blockIdx.x;
    if (threadIdx.x == 0) {
        int kept = 0;
        #pragma unroll
        for (int j = 0; j < NUM_BEAMS; ++j) { selTok[j] = 0; selSc[j] = -3.4e38f; }
        for (int j = 0; j < NUM_BEAMS + 1 && kept < NUM_BEAMS; ++j) {
            int tok = topi[b * 8 + j];
            if (tok != EOS) { selTok[kept] = tok; selSc[kept] = topv[b * 8 + j]; ++kept; }
        }
    }
    __syncthreads();
    const int i = threadIdx.x;      // 128 threads = 4 beams x 32 cols
    const int j = i >> 5, c = i & 31;
    const int row = b * NUM_BEAMS + j;
    int val = 0;                    // PAD
    if (c == 0) val = tokens0[b];
    if (c == 1) val = selTok[j];
    tokOut[row * MAX_LEN + c] = val;
    if (c == 0) beamScores[row] = selSc[j];
}

// ---------------------------------------------------------------
// Beam step: merge 4 sorted 8-lists -> global top-8 (ties favor lower
// beam, matching flat-index order), keep first 4 non-EOS, reorder rows.
// ---------------------------------------------------------------
__global__ void beam_step_kernel(const float* __restrict__ topv,
                                 const int* __restrict__ topi,
                                 const int* __restrict__ tokIn,
                                 int* __restrict__ tokOut,
                                 float* __restrict__ beamScores, int t) {
    __shared__ int   selTok[NUM_BEAMS], selBeam[NUM_BEAMS];
    __shared__ float selSc[NUM_BEAMS];
    const int b = blockIdx.x;
    if (threadIdx.x == 0) {
        int ptr[NUM_BEAMS] = {0, 0, 0, 0};
        #pragma unroll
        for (int j = 0; j < NUM_BEAMS; ++j) { selTok[j] = 0; selBeam[j] = 0; selSc[j] = -3.4e38f; }
        int kept = 0, scanned = 0;
        while (kept < NUM_BEAMS && scanned < 2 * NUM_BEAMS) {
            int bestBeam = 0; float bestV = -3.4e38f;
            for (int k = 0; k < NUM_BEAMS; ++k) {
                if (ptr[k] < 8) {
                    float v = topv[(b * NUM_BEAMS + k) * 8 + ptr[k]];
                    if (v > bestV) { bestV = v; bestBeam = k; }
                }
            }
            int tok = topi[(b * NUM_BEAMS + bestBeam) * 8 + ptr[bestBeam]];
            ptr[bestBeam]++;
            scanned++;
            if (tok != EOS) {
                selTok[kept] = tok; selBeam[kept] = bestBeam; selSc[kept] = bestV; ++kept;
            }
        }
    }
    __syncthreads();
    const int i = threadIdx.x;      // 128 threads = 4 beams x 32 cols
    const int j = i >> 5, c = i & 31;
    const int srcRow = b * NUM_BEAMS + selBeam[j];
    const int dstRow = b * NUM_BEAMS + j;
    int val = (c == t) ? selTok[j] : tokIn[srcRow * MAX_LEN + c];
    tokOut[dstRow * MAX_LEN + c] = val;
    if (c == 0) beamScores[dstRow] = selSc[j];
}

// ---------------------------------------------------------------
// Emit outputs: token ids (as float) then beam scores.
// ---------------------------------------------------------------
__global__ void write_out_kernel(const int* __restrict__ tokFinal,
                                 const float* __restrict__ beamScores,
                                 float* __restrict__ out, int outSize) {
    int i = blockIdx.x * blockDim.x + threadIdx.x;
    if (i >= outSize) return;
    const int nTok = ROWS_MAX * MAX_LEN;
    if (i < nTok)                     out[i] = (float)tokFinal[i];
    else if (i < nTok + ROWS_MAX)     out[i] = beamScores[i - nTok];
}

// ---------------------------------------------------------------
extern "C" void kernel_launch(void* const* d_in, const int* in_sizes, int n_in,
                              void* d_out, int out_size, void* d_ws, size_t ws_size,
                              hipStream_t stream) {
    const float* emb    = (const float*)d_in[0];   // [VOCAB, 512]
    const float* proj   = (const float*)d_in[1];   // [512, VOCAB]
    const int*   tokens = (const int*)d_in[2];     // [BATCH, 1]

    char*  ws  = (char*)d_ws;
    size_t off = 0;
    auto carve = [&](size_t bytes) -> void* {
        void* p = ws + off;
        off = (off + bytes + 255) & ~(size_t)255;
        return p;
    };
    bf16*  projp  = (bf16*)carve((size_t)D_MODEL * VPAD * sizeof(bf16));    // 51.9 MB (L2-resident)
    float* logits = (float*)carve((size_t)ROWS_MAX * VPAD * sizeof(float)); // 26.0 MB
    bf16*  hbf    = (bf16*)carve((size_t)ROWS_MAX * D_MODEL * sizeof(bf16));
    float* topv   = (float*)carve(ROWS_MAX * 8 * sizeof(float));
    int*   topi   = (int*)carve(ROWS_MAX * 8 * sizeof(int));
    int*   tokA   = (int*)carve(ROWS_MAX * MAX_LEN * sizeof(int));
    int*   tokB   = (int*)carve(ROWS_MAX * MAX_LEN * sizeof(int));
    float* bsc    = (float*)carve(ROWS_MAX * sizeof(float));

    // ---- one-time (per call) bf16 tile pack of proj ----
    {
        long total = (long)D_MODEL * VPAD;
        pack_proj_kernel<<<(int)((total + 255) / 256), 256, 0, stream>>>(proj, projp);
    }

    // ---- first decode step (rows = BATCH, cur_len = 1) ----
    pool_embed_kernel<<<BATCH, 128, 0, stream>>>(emb, tokens, 1, 1, hbf);
    gemm_wmma_kernel<<<dim3(NTILES / BLK_NT, BATCH / 16), 256, 0, stream>>>(hbf, projp, logits);
    row_softtop_kernel<<<BATCH, 256, 0, stream>>>(logits, nullptr, topv, topi);
    beam_init_kernel<<<BATCH, 128, 0, stream>>>(topv, topi, tokens, tokA, bsc);

    // ---- main loop: t = 2 .. MAX_LEN-1 ----
    int* cur = tokA;
    int* nxt = tokB;
    for (int t = 2; t < MAX_LEN; ++t) {
        pool_embed_kernel<<<ROWS_MAX, 128, 0, stream>>>(emb, cur, MAX_LEN, t, hbf);
        gemm_wmma_kernel<<<dim3(NTILES / BLK_NT, ROWS_MAX / 16), 256, 0, stream>>>(hbf, projp, logits);
        rep_penalty_kernel<<<1, ROWS_MAX, 0, stream>>>(logits, cur, t, ROWS_MAX,
                                                       (t + 1 == MAX_LEN) ? 1 : 0);
        row_softtop_kernel<<<ROWS_MAX, 256, 0, stream>>>(logits, bsc, topv, topi);
        beam_step_kernel<<<BATCH, 128, 0, stream>>>(topv, topi, cur, nxt, bsc, t);
        int* tmp = cur; cur = nxt; nxt = tmp;
    }

    // ---- outputs: [128*32] token ids (as float) + [128] beam scores ----
    int total = ROWS_MAX * MAX_LEN + ROWS_MAX;
    if (total > out_size) total = out_size;
    write_out_kernel<<<(total + 255) / 256, 256, 0, stream>>>(cur, bsc, (float*)d_out, total);
}
